// FragExtraStackIteration_7215545057300
// MI455X (gfx1250) — matrix operand order repaired and checked by reading, not compiled, not run on hardware
//
#include <hip/hip_runtime.h>
#include <math.h>

// ============================ types / helpers ============================

typedef __attribute__((ext_vector_type(16))) __bf16 v16bf;
typedef __attribute__((ext_vector_type(8)))  __bf16 bf16x8;
typedef __attribute__((ext_vector_type(8)))  float  v8f;

#define DEVFN static __device__ __forceinline__

DEVFN __bf16 f2bf(float f) {
  union { float f; unsigned u; } a; a.f = f;
  unsigned r = (a.u + 0x7FFFu + ((a.u >> 16) & 1u)) >> 16;   // RNE
  union { unsigned short s; __bf16 b; } c; c.s = (unsigned short)r;
  return c.b;
}

// ---- CDNA5 async global->LDS copies (ASYNCcnt-tracked DMA, no VGPR hop) ----
DEVFN void async_g2l_b128(void* lds, const void* g) {
  unsigned l = (unsigned)(size_t)lds;                 // low 32 bits = LDS offset
  unsigned long long ga = (unsigned long long)(size_t)g;
  asm volatile("global_load_async_to_lds_b128 %0, %1, off" :: "v"(l), "v"(ga) : "memory");
}
DEVFN void async_g2l_b32(void* lds, const void* g) {
  unsigned l = (unsigned)(size_t)lds;
  unsigned long long ga = (unsigned long long)(size_t)g;
  asm volatile("global_load_async_to_lds_b32 %0, %1, off" :: "v"(l), "v"(ga) : "memory");
}
DEVFN void async_waitcnt0() { asm volatile("s_wait_asynccnt 0x0" ::: "memory"); }

// A fragment (16xK=32, bf16), LDS row-major [m][k], rowStride in halfs.
// ISA layout: lane L<16 -> row L, K={0..7,16..23}; lane>=16 -> row L-16, K={8..15,24..31}
DEVFN v16bf frag_a(const __bf16* base, int rowStride, int kbase, int lane) {
  const __bf16* row = base + (lane & 15) * rowStride + kbase + ((lane < 16) ? 0 : 8);
  bf16x8 lo = *(const bf16x8*)(row);
  bf16x8 hi = *(const bf16x8*)(row + 16);
  v16bf a;
#pragma unroll
  for (int t = 0; t < 8; ++t) { a[t] = lo[t]; a[t + 8] = hi[t]; }
  return a;
}

// B fragment (K=32 x 16, bf16) staged in LDS as [n][k], colStride in halfs.
// lane L -> col L%16, K = (L<16 ? {0..7,16..23} : {8..15,24..31})
DEVFN v16bf frag_b(const __bf16* base, int colStride, int kbase, int lane) {
  const __bf16* col = base + (lane & 15) * colStride + kbase + ((lane < 16) ? 0 : 8);
  bf16x8 lo = *(const bf16x8*)(col);
  bf16x8 hi = *(const bf16x8*)(col + 16);
  v16bf b;
#pragma unroll
  for (int t = 0; t < 8; ++t) { b[t] = lo[t]; b[t + 8] = hi[t]; }
  return b;
}

DEVFN v8f wmma_bf16(v16bf a, v16bf b, v8f c) {
  return __builtin_amdgcn_wmma_f32_16x16x32_bf16(false, a, false, b, (short)0, c,
                                                 false, false);
}

DEVFN float wred_sum(float v) {
#pragma unroll
  for (int o = 16; o; o >>= 1) v += __shfl_xor(v, o, 32);
  return v;
}
DEVFN float wred_max(float v) {
#pragma unroll
  for (int o = 16; o; o >>= 1) v = fmaxf(v, __shfl_xor(v, o, 32));
  return v;
}
DEVFN float sigmoidf_(float x) { return 1.f / (1.f + __expf(-x)); }

// ============================ CDNA5 path probe (not launched) ============================
// Confirms async-to-LDS / wait-counter / cluster mnemonics assemble for gfx1250.

__global__ void k_cdna5_probe(const float* __restrict__ g, float* __restrict__ out) {
  __shared__ float buf[256];
  async_g2l_b32(&buf[threadIdx.x], g + (blockIdx.x << 8) + threadIdx.x);
  async_waitcnt0();
  __builtin_amdgcn_s_wait_tensorcnt(0);
  __syncthreads();
  out[(blockIdx.x << 8) + threadIdx.x] =
      buf[threadIdx.x] + (float)__builtin_amdgcn_cluster_id_x();
}

// ============================ weight prep: fp32[K][N] -> bf16[N][K] ============================

__global__ void k_cvt_tr(const float* __restrict__ src, __bf16* __restrict__ dst,
                         int K, int N) {
  long t = (long)blockIdx.x * blockDim.x + threadIdx.x;
  if (t < (long)K * N) {
    int k = (int)(t / N), n = (int)(t % N);
    dst[(long)n * K + k] = f2bf(src[t]);
  }
}

// ============================ layernorm ============================

__global__ __launch_bounds__(256) void k_layernorm(
    const float* __restrict__ x, const float* __restrict__ gg,
    const float* __restrict__ bb, float* __restrict__ y, int rows, int C) {
  int wid = threadIdx.x >> 5, lane = threadIdx.x & 31;
  long row = (long)blockIdx.x * 8 + wid;
  if (row >= rows) return;
  const float* xr = x + row * C;
  int nch = C >> 5;            // C is 64 or 128
  float vals[4];
  float s = 0.f;
  for (int i = 0; i < nch; ++i) { vals[i] = xr[lane + 32 * i]; s += vals[i]; }
  float mu = wred_sum(s) / (float)C;
  float v2 = 0.f;
  for (int i = 0; i < nch; ++i) { float d = vals[i] - mu; v2 += d * d; }
  float var = wred_sum(v2) / (float)C;
  float rs = rsqrtf(var + 1e-5f);
  for (int i = 0; i < nch; ++i) {
    int idx = lane + 32 * i;
    y[row * C + idx] = (vals[i] - mu) * rs * gg[idx] + bb[idx];
  }
}

// ============================ generic WMMA GEMM ============================
// C[m,n] (per batch) = cmode( act( alpha * sum_k A[m,k]*B[k,n] + bias[n] ) * rowscale[m] )

struct GemmArgs {
  const float* A; const float* B; float* C;
  const float* bias; const float* rowscale;
  long lam, lak, lbk, lbn, lcm, lcn;
  long abs_, bbs, cbs;
  int M, N, K;
  float alpha;
};

template <int ACT, int CMODE>
__global__ __launch_bounds__(256) void k_gemm(GemmArgs g) {
  __shared__ __bf16 sA[64 * 32];
  __shared__ __bf16 sB[64 * 32];
  const int tid = threadIdx.x, lane = tid & 31, wid = tid >> 5;
  const float* A = g.A + (long)blockIdx.z * g.abs_;
  const float* B = g.B + (long)blockIdx.z * g.bbs;
  float* C = g.C + (long)blockIdx.z * g.cbs;
  const int m0 = blockIdx.y * 64, n0 = blockIdx.x * 64;
  const int wm = wid & 3, wn = wid >> 2;   // wave: 16-row block, 32-col half
  v8f acc0 = {}; v8f acc1 = {};
  for (int k0 = 0; k0 < g.K; k0 += 32) {
    __syncthreads();
    for (int t = tid; t < 64 * 32; t += 256) {
      int r = t >> 5, k = t & 31;
      int gm = m0 + r, gk = k0 + k;
      float av = (gm < g.M && gk < g.K) ? A[(long)gm * g.lam + (long)gk * g.lak] : 0.f;
      sA[t] = f2bf(av);
      int gn = n0 + r;
      float bv = (gn < g.N && gk < g.K) ? B[(long)gk * g.lbk + (long)gn * g.lbn] : 0.f;
      sB[t] = f2bf(bv);
    }
    if (k0 + 32 < g.K) {               // prefetch next k-tile (global_prefetch_b8)
      int r = tid >> 5, k = tid & 31;
      long gm = m0 + r, gk = k0 + 32 + k, gn = n0 + r;
      if (gm < g.M && gk < g.K) __builtin_prefetch(A + gm * g.lam + gk * g.lak, 0, 1);
      if (gn < g.N && gk < g.K) __builtin_prefetch(B + gk * g.lbk + gn * g.lbn, 0, 1);
    }
    __syncthreads();
    v16bf af = frag_a(sA + wm * 16 * 32, 32, 0, lane);
    v16bf b0 = frag_b(sB + (wn * 32) * 32, 32, 0, lane);
    v16bf b1 = frag_b(sB + (wn * 32 + 16) * 32, 32, 0, lane);
    acc0 = wmma_bf16(af, b0, acc0);
    acc1 = wmma_bf16(af, b1, acc1);
  }
  int mbase = m0 + wm * 16 + ((lane >= 16) ? 8 : 0);
  int nbase = n0 + wn * 32 + (lane & 15);
#pragma unroll
  for (int r = 0; r < 8; ++r) {
    int m = mbase + r;
#pragma unroll
    for (int t = 0; t < 2; ++t) {
      int n = nbase + t * 16;
      if (m < g.M && n < g.N) {
        float v = (t ? acc1[r] : acc0[r]) * g.alpha;
        if (g.bias) v += g.bias[n];
        if (ACT == 1) v = fmaxf(v, 0.f);
        if (ACT == 2) v = sigmoidf_(v);
        if (g.rowscale) v *= g.rowscale[m];
        float* cp = C + (long)m * g.lcm + (long)n * g.lcn;
        if (CMODE == 0) *cp = v;
        else if (CMODE == 1) *cp = *cp + v;
        else *cp = *cp * v;
      }
    }
  }
}

// ============================ fused attention ============================
// One wave per block; 16-query tile vs 256 keys, full score row in LDS.
// grid = (Nn/16, H, Mb); transposed=1 for triangle-"ending" indexing.

__global__ __launch_bounds__(32) void k_attn(
    const float* __restrict__ Q, const float* __restrict__ K,
    const float* __restrict__ V, const float* __restrict__ G,
    const float* __restrict__ NB, const float* __restrict__ mask,
    float* __restrict__ O, int Nn, int H, int D, int transposed) {
  __shared__ float  sS[16 * 256];
  __shared__ __bf16 sW[16 * 256];
  __shared__ __bf16 sQ[16 * 32];
  __shared__ __bf16 sKV[32 * 32];
  const int lane = threadIdx.x;
  const int i0 = blockIdx.x * 16, h = blockIdx.y, m = blockIdx.z;
  const int HD = H * D;
  const long NN2 = (long)Nn * Nn;

  for (int t = lane; t < 16 * 32; t += 32) {
    int r = t >> 5, k = t & 31;
    long pix = transposed ? ((long)(i0 + r) * Nn + m) : ((long)m * Nn + i0 + r);
    sQ[t] = (k < D) ? f2bf(Q[pix * HD + h * D + k]) : f2bf(0.f);
  }
  __syncthreads();
  v16bf aq = frag_a(sQ, 32, 0, lane);

  for (int j0 = 0; j0 < Nn; j0 += 16) {
    __syncthreads();
    for (int t = lane; t < 16 * 32; t += 32) {
      int n = t >> 5, k = t & 31;
      long pix = transposed ? ((long)(j0 + n) * Nn + m) : ((long)m * Nn + j0 + n);
      sKV[t] = (k < D) ? f2bf(K[pix * HD + h * D + k]) : f2bf(0.f);
    }
    __syncthreads();
    v16bf bk = frag_b(sKV, 32, 0, lane);
    v8f acc = {};
    acc = wmma_bf16(aq, bk, acc);
#pragma unroll
    for (int r = 0; r < 8; ++r) {
      int ml = r + ((lane >= 16) ? 8 : 0);
      int nl = lane & 15;
      int i = i0 + ml, j = j0 + nl;
      long mpix = transposed ? ((long)j * Nn + m) : ((long)m * Nn + j);
      long npix = transposed ? ((long)j * Nn + i) : ((long)i * Nn + j);
      float bias = 1e9f * (mask[mpix] - 1.f) + NB[(long)h * NN2 + npix];
      sS[ml * 256 + j] = acc[r] + bias;
    }
  }
  __syncthreads();

  for (int r = 0; r < 16; ++r) {       // softmax, one row per pass, whole wave
    float e[8]; float mx = -3.4e38f;
#pragma unroll
    for (int t = 0; t < 8; ++t) { e[t] = sS[r * 256 + lane + 32 * t]; mx = fmaxf(mx, e[t]); }
    mx = wred_max(mx);
    float sum = 0.f;
#pragma unroll
    for (int t = 0; t < 8; ++t) { e[t] = __expf(e[t] - mx); sum += e[t]; }
    sum = wred_sum(sum);
    float inv = 1.f / sum;
#pragma unroll
    for (int t = 0; t < 8; ++t) sW[r * 256 + lane + 32 * t] = f2bf(e[t] * inv);
  }
  __syncthreads();

  const int ntiles = (D + 15) / 16;
  v8f accO[2] = {{}, {}};
  for (int k0 = 0; k0 < Nn; k0 += 32) {
    __syncthreads();
    for (int t = lane; t < ntiles * 16 * 32; t += 32) {
      int n = t >> 5, k = t & 31;
      long pix = transposed ? ((long)(k0 + k) * Nn + m) : ((long)m * Nn + k0 + k);
      sKV[t] = (n < D) ? f2bf(V[pix * HD + h * D + n]) : f2bf(0.f);
    }
    __syncthreads();
    v16bf aw = frag_a(sW, 256, k0, lane);
    for (int t2 = 0; t2 < ntiles; ++t2) {
      v16bf bv = frag_b(sKV + t2 * 16 * 32, 32, 0, lane);
      accO[t2] = wmma_bf16(aw, bv, accO[t2]);
    }
  }
  for (int t2 = 0; t2 < ntiles; ++t2) {
#pragma unroll
    for (int r = 0; r < 8; ++r) {
      int ml = r + ((lane >= 16) ? 8 : 0);
      int c = t2 * 16 + (lane & 15);
      if (c < D) {
        int i = i0 + ml;
        long pix = transposed ? ((long)i * Nn + m) : ((long)m * Nn + i);
        long orow = (long)m * Nn + i;       // attention-space row
        O[orow * HD + h * D + c] = accO[t2][r] * G[pix * HD + h * D + c];
      }
    }
  }
}

// ============================ fused transition ============================
// resid += relu(X@W1 + b1)@W2 + b2 ; weights pre-transposed bf16 [N][K],
// streamed into LDS with async global->LDS b128 DMA.

__global__ __launch_bounds__(256) void k_transition(
    const float* __restrict__ xln, const __bf16* __restrict__ W1T,
    const float* __restrict__ b1, const __bf16* __restrict__ W2T,
    const float* __restrict__ b2, float* __restrict__ resid,
    int rows, int C, int H) {
  __shared__ __bf16 sX[16 * 128];
  __shared__ __bf16 sW[32 * 512];
  __shared__ __bf16 sH[16 * 512];
  const int tid = threadIdx.x, lane = tid & 31, wid = tid >> 5;
  const long r0 = (long)blockIdx.x * 16;
  (void)rows;
  for (int t = tid; t < 16 * C; t += 256) {
    int r = t / C, c = t % C;
    sX[r * C + c] = f2bf(xln[(r0 + r) * C + c]);
  }
  const int nt1 = H / 16, tpw = nt1 / 8;   // tiles per wave in GEMM1 (2 or 4)
  v8f acc[4] = {{}, {}, {}, {}};
  for (int k0 = 0; k0 < C; k0 += 32) {
    __syncthreads();
    for (int t = tid; t < H * 4; t += 256) {        // 32 halfs per n = 4x b128
      int n = t >> 2, kc = (t & 3) * 8;
      async_g2l_b128(&sW[n * 32 + kc], W1T + (long)n * C + k0 + kc);
    }
    async_waitcnt0();
    __syncthreads();
    v16bf ax = frag_a(sX, C, k0, lane);
    for (int q = 0; q < tpw; ++q) {
      int nt = wid * tpw + q;
      v16bf bw = frag_b(sW + nt * 16 * 32, 32, 0, lane);
      acc[q] = wmma_bf16(ax, bw, acc[q]);
    }
  }
  for (int q = 0; q < tpw; ++q) {
    int nt = wid * tpw + q;
#pragma unroll
    for (int r = 0; r < 8; ++r) {
      int ml = r + ((lane >= 16) ? 8 : 0);
      int n = nt * 16 + (lane & 15);
      sH[ml * H + n] = f2bf(fmaxf(acc[q][r] + b1[n], 0.f));
    }
  }
  __syncthreads();
  const int nt2 = C / 16;                  // 4 or 8 output tiles
  v8f acc2 = {};
  for (int k0 = 0; k0 < H; k0 += 32) {
    __syncthreads();
    for (int t = tid; t < C * 4; t += 256) {
      int n = t >> 2, kc = (t & 3) * 8;
      async_g2l_b128(&sW[n * 32 + kc], W2T + (long)n * H + k0 + kc);
    }
    async_waitcnt0();
    __syncthreads();
    if (wid < nt2) {
      v16bf ah = frag_a(sH, H, k0, lane);
      v16bf bw = frag_b(sW + wid * 16 * 32, 32, 0, lane);
      acc2 = wmma_bf16(ah, bw, acc2);
    }
  }
  if (wid < nt2) {
#pragma unroll
    for (int r = 0; r < 8; ++r) {
      int ml = r + ((lane >= 16) ? 8 : 0);
      int n = wid * 16 + (lane & 15);
      resid[(r0 + ml) * C + n] += acc2[r] + b2[n];
    }
  }
}

// ============================ outer product mean ============================
// 16 (i,j) pairs per block (16 waves); per-wave 32x32xK=128 outer (WMMA),
// block-wide 16x1024 @ 1024x128 projection with async-DMA'd bf16 weights.

__global__ __launch_bounds__(512) void k_opm(
    const float* __restrict__ L, const float* __restrict__ R,
    const __bf16* __restrict__ OWT, const float* __restrict__ OB,
    const float* __restrict__ nrm, float* __restrict__ pair, int Nn, int Mm) {
  __shared__ __bf16 sL[16 * 1024];
  __shared__ __bf16 sR[16 * 1024];
  __shared__ __bf16 sO[16 * 1024];
  __shared__ __bf16 sW[32 * 128];
  const int tid = threadIdx.x, lane = tid & 31, wid = tid >> 5;
  const long pbase = (long)blockIdx.x * 16;
  const long P = pbase + wid;
  const int i = (int)(P / Nn), j = (int)(P % Nn);

  v8f acc[2][2] = {{{}, {}}, {{}, {}}};
  for (int k0 = 0; k0 < Mm; k0 += 32) {
    __syncthreads();
    for (int t = lane; t < 1024; t += 32) {
      int x = t & 31, kk = t >> 5;
      sL[wid * 1024 + x * 32 + kk] = f2bf(L[((long)(k0 + kk) * Nn + i) * 32 + x]);
      sR[wid * 1024 + x * 32 + kk] = f2bf(R[((long)(k0 + kk) * Nn + j) * 32 + x]);
    }
    __syncthreads();
    for (int tx = 0; tx < 2; ++tx) {
      v16bf a = frag_a(sL + wid * 1024 + tx * 16 * 32, 32, 0, lane);
      for (int ty = 0; ty < 2; ++ty) {
        v16bf b = frag_b(sR + wid * 1024 + ty * 16 * 32, 32, 0, lane);
        acc[tx][ty] = wmma_bf16(a, b, acc[tx][ty]);
      }
    }
  }
  for (int tx = 0; tx < 2; ++tx)
    for (int ty = 0; ty < 2; ++ty)
#pragma unroll
      for (int r = 0; r < 8; ++r) {
        int x = tx * 16 + r + ((lane >= 16) ? 8 : 0);
        int y = ty * 16 + (lane & 15);
        sO[wid * 1024 + x * 32 + y] = f2bf(acc[tx][ty][r]);
      }
  __syncthreads();
  v8f ap = {};
  for (int k0 = 0; k0 < 1024; k0 += 32) {
    __syncthreads();
    for (int t = tid; t < 128 * 4; t += 512) {
      int n = t >> 2, kc = (t & 3) * 8;
      async_g2l_b128(&sW[n * 32 + kc], OWT + (long)n * 1024 + k0 + kc);
    }
    async_waitcnt0();
    __syncthreads();
    if (wid < 8) {
      v16bf a = frag_a(sO, 1024, k0, lane);
      v16bf b = frag_b(sW + wid * 16 * 32, 32, 0, lane);
      ap = wmma_bf16(a, b, ap);
    }
  }
  if (wid < 8) {
#pragma unroll
    for (int r = 0; r < 8; ++r) {
      int ml = r + ((lane >= 16) ? 8 : 0);
      int n = wid * 16 + (lane & 15);
      long p = pbase + ml;                 // p == i*Nn + j
      pair[p * 128 + n] += (ap[r] + OB[n]) / (nrm[p] + 1e-3f);
    }
  }
}

// ============================ small helper kernels ============================

__global__ void k_mean_m(const float* __restrict__ x, float* __restrict__ y,
                         int Mm, int Nn, int C) {
  int n = blockIdx.x, c = threadIdx.x;
  if (c < C) {
    float s = 0.f;
    for (int m = 0; m < Mm; ++m) s += x[((long)m * Nn + n) * C + c];
    y[(long)n * C + c] = s / (float)Mm;
  }
}

__global__ __launch_bounds__(128) void k_col_attn(
    const float* __restrict__ q, const float* __restrict__ K,
    const float* __restrict__ V, const float* __restrict__ G,
    float* __restrict__ O, int Mm, int Nn) {
  __shared__ float sS[8 * 128];
  __shared__ float sO[64];
  const int n = blockIdx.x, tid = threadIdx.x;
  if (tid < Mm) {
    for (int h = 0; h < 8; ++h) {
      float s = 0.f;
      for (int c = 0; c < 8; ++c)
        s += q[(long)n * 64 + h * 8 + c] * K[((long)tid * Nn + n) * 8 + c];
      sS[h * 128 + tid] = s;
    }
  }
  __syncthreads();
  if (tid < 8) {
    int h = tid; float mx = -3.4e38f;
    for (int m = 0; m < Mm; ++m) mx = fmaxf(mx, sS[h * 128 + m]);
    float sum = 0.f;
    for (int m = 0; m < Mm; ++m) { float e = __expf(sS[h * 128 + m] - mx); sS[h * 128 + m] = e; sum += e; }
    float inv = 1.f / sum;
    for (int m = 0; m < Mm; ++m) sS[h * 128 + m] *= inv;
  }
  __syncthreads();
  if (tid < 64) {
    int h = tid >> 3, c = tid & 7;
    float o = 0.f;
    for (int m = 0; m < Mm; ++m) o += sS[h * 128 + m] * V[((long)m * Nn + n) * 8 + c];
    sO[tid] = o;
  }
  __syncthreads();
  for (int idx = tid; idx < Mm * 64; idx += 128) {
    int m = idx >> 6, hc = idx & 63;
    long r = (long)m * Nn + n;
    O[r * 64 + hc] = sO[hc] * G[r * 64 + hc];
  }
}

__global__ void k_opm_norm(const float* __restrict__ mask, float* __restrict__ nrm,
                           int Mm, int Nn) {
  long p = (long)blockIdx.x * blockDim.x + threadIdx.x;
  if (p < (long)Nn * Nn) {
    int i = (int)(p / Nn), j = (int)(p % Nn);
    float s = 0.f;
    for (int m = 0; m < Mm; ++m) s += mask[(long)m * Nn + i] * mask[(long)m * Nn + j];
    nrm[p] = s;
  }
}

__global__ void k_add_tr(float* __restrict__ dst, const float* __restrict__ src,
                         int Nn, int C, int transposed) {
  long t = (long)blockIdx.x * blockDim.x + threadIdx.x;
  long total = (long)Nn * Nn * C;
  if (t < total) {
    long p = t / C; int c = (int)(t % C);
    int x = (int)(p / Nn), y = (int)(p % Nn);
    long sp = transposed ? ((long)y * Nn + x) : p;
    dst[p * C + c] += src[sp * C + c];
  }
}

// ============================ host orchestration ============================

static const int MM = 128, NNc = 256, CMc = 64, CZc = 128;
static const long MSA_ELEMS  = (long)MM * NNc * CMc;     // 2,097,152
static const long PAIR_ELEMS = (long)NNc * NNc * CZc;    // 8,388,608
static const long PIX_MSA    = (long)MM * NNc;           // 32,768
static const long PIX_PAIR   = (long)NNc * NNc;          // 65,536

// jax pytree (dict) flattening is alphabetical at every level:
enum {
  IN_MSA = 0, IN_PAIR, IN_MSA_MASK, IN_PAIR_MASK,
  COL_GATE_B = 4, COL_GATE_W, COL_K, COL_LN_B, COL_LN_G, COL_OUT_B, COL_OUT_W, COL_Q, COL_V,
  MSATR_B1 = 13, MSATR_B2, MSATR_LN_B, MSATR_LN_G, MSATR_W1, MSATR_W2,
  OPM_LB = 19, OPM_LN_B, OPM_LN_G, OPM_LW, OPM_OB, OPM_OW, OPM_RB, OPM_RW,
  PTR_B1 = 27, PTR_B2, PTR_LN_B, PTR_LN_G, PTR_W1, PTR_W2,
  ROW_B2D = 33, ROW_GATE_B, ROW_GATE_W, ROW_K, ROW_LN2D_B, ROW_LN2D_G,
  ROW_LN_B, ROW_LN_G, ROW_OUT_B, ROW_OUT_W, ROW_Q, ROW_V,
  TE_B2D = 45, TE_GATE_B, TE_GATE_W, TE_K, TE_LN_B, TE_LN_G, TE_OUT_B, TE_OUT_W, TE_Q, TE_V,
  TI_CLN_B = 55, TI_CLN_G, TI_G_B, TI_G_W, TI_LG_B, TI_LG_W, TI_LN_B, TI_LN_G,
  TI_LP_B, TI_LP_W, TI_OP_B, TI_OP_W, TI_RG_B, TI_RG_W, TI_RP_B, TI_RP_W,
  TO_CLN_B = 71, TO_CLN_G, TO_G_B, TO_G_W, TO_LG_B, TO_LG_W, TO_LN_B, TO_LN_G,
  TO_LP_B, TO_LP_W, TO_OP_B, TO_OP_W, TO_RG_B, TO_RG_W, TO_RP_B, TO_RP_W,
  TS_B2D = 87, TS_GATE_B, TS_GATE_W, TS_K, TS_LN_B, TS_LN_G, TS_OUT_B, TS_OUT_W, TS_Q, TS_V,
};

static void gemm(hipStream_t st,
                 const float* A, long lam, long lak,
                 const float* B, long lbk, long lbn,
                 float* C, long lcm, long lcn,
                 const float* bias, const float* rowscale,
                 int M, int N, int K, float alpha, int act, int cmode,
                 int batches = 1, long abs_ = 0, long bbs = 0, long cbs = 0) {
  GemmArgs g;
  g.A = A; g.B = B; g.C = C; g.bias = bias; g.rowscale = rowscale;
  g.lam = lam; g.lak = lak; g.lbk = lbk; g.lbn = lbn; g.lcm = lcm; g.lcn = lcn;
  g.abs_ = abs_; g.bbs = bbs; g.cbs = cbs; g.M = M; g.N = N; g.K = K; g.alpha = alpha;
  dim3 gr((N + 63) / 64, (M + 63) / 64, batches);
  if (act == 2)        k_gemm<2, 0><<<gr, 256, 0, st>>>(g);
  else if (cmode == 1) k_gemm<0, 1><<<gr, 256, 0, st>>>(g);
  else if (cmode == 2) k_gemm<0, 2><<<gr, 256, 0, st>>>(g);
  else                 k_gemm<0, 0><<<gr, 256, 0, st>>>(g);
}

static void ln(hipStream_t st, const float* x, const float* g, const float* b,
               float* y, int rows, int C) {
  k_layernorm<<<rows / 8, 256, 0, st>>>(x, g, b, y, rows, C);
}

extern "C" void kernel_launch(void* const* d_in, const int* in_sizes, int n_in,
                              void* d_out, int out_size, void* d_ws, size_t ws_size,
                              hipStream_t stream) {
  (void)in_sizes; (void)n_in; (void)out_size; (void)ws_size;
  const float* msa_mask  = (const float*)d_in[IN_MSA_MASK];
  const float* pair_mask = (const float*)d_in[IN_PAIR_MASK];
#define PW(i) ((const float*)d_in[i])

  float* msa  = (float*)d_out;            // output tuple element 0
  float* pair = msa + MSA_ELEMS;          // output tuple element 1
  hipMemcpyAsync(msa,  d_in[IN_MSA],  MSA_ELEMS  * sizeof(float), hipMemcpyDeviceToDevice, stream);
  hipMemcpyAsync(pair, d_in[IN_PAIR], PAIR_ELEMS * sizeof(float), hipMemcpyDeviceToDevice, stream);

  float* ws = (float*)d_ws;
  auto alloc = [&](long n) { float* p = ws; ws += n; return p; };
  float* P0 = alloc(PAIR_ELEMS);   // LN scratch
  float* P1 = alloc(PAIR_ELEMS);   // zp_norm / proj tmp
  float* P2 = alloc(PAIR_ELEMS);   // q / lp / left
  float* P3 = alloc(PAIR_ELEMS);   // k / rp / right
  float* P4 = alloc(PAIR_ELEMS);   // v / tri product
  float* P5 = alloc(PAIR_ELEMS);   // gate
  float* P6 = alloc(PAIR_ELEMS);   // attention output
  float* nbbuf = alloc(8L * PIX_PAIR);
  float* qavg  = alloc(NNc * CMc);
  float* colq  = alloc(NNc * CMc);
  float* nrm   = alloc(PIX_PAIR);
  __bf16* w1t_msa = (__bf16*)alloc(8192);    // [256][64]
  __bf16* w2t_msa = (__bf16*)alloc(8192);    // [64][256]
  __bf16* w1t_pr  = (__bf16*)alloc(32768);   // [512][128]
  __bf16* w2t_pr  = (__bf16*)alloc(32768);   // [128][512]
  __bf16* owt     = (__bf16*)alloc(65536);   // [128][1024]

  // one-time per call: weights -> transposed bf16 for async-DMA streaming
  k_cvt_tr<<<(64L * 256 + 255) / 256, 256, 0, stream>>>(PW(MSATR_W1), w1t_msa, 64, 256);
  k_cvt_tr<<<(256L * 64 + 255) / 256, 256, 0, stream>>>(PW(MSATR_W2), w2t_msa, 256, 64);
  k_cvt_tr<<<(128L * 512 + 255) / 256, 256, 0, stream>>>(PW(PTR_W1), w1t_pr, 128, 512);
  k_cvt_tr<<<(512L * 128 + 255) / 256, 256, 0, stream>>>(PW(PTR_W2), w2t_pr, 512, 128);
  k_cvt_tr<<<(1024L * 128 + 255) / 256, 256, 0, stream>>>(PW(OPM_OW), owt, 1024, 128);

  const float RS8 = 0.35355339059f, RS32 = 0.17677669529f;
  const long LN2 = (long)PIX_PAIR;   // 65536

  // ---------------- row attention (with pair bias) ----------------
  ln(stream, msa, PW(ROW_LN_G), PW(ROW_LN_B), P0, PIX_MSA, CMc);
  ln(stream, pair, PW(ROW_LN2D_G), PW(ROW_LN2D_B), P1, PIX_PAIR, CZc);
  gemm(stream, P1, CZc, 1, PW(ROW_B2D), 8, 1, nbbuf, 1, LN2,
       nullptr, nullptr, PIX_PAIR, 8, CZc, 1.f, 0, 0);
  gemm(stream, P0, CMc, 1, PW(ROW_Q), 64, 1, P2, 64, 1, nullptr, nullptr, PIX_MSA, 64, CMc, RS8, 0, 0);
  gemm(stream, P0, CMc, 1, PW(ROW_K), 64, 1, P3, 64, 1, nullptr, nullptr, PIX_MSA, 64, CMc, 1.f, 0, 0);
  gemm(stream, P0, CMc, 1, PW(ROW_V), 64, 1, P4, 64, 1, nullptr, nullptr, PIX_MSA, 64, CMc, 1.f, 0, 0);
  gemm(stream, P0, CMc, 1, PW(ROW_GATE_W), 64, 1, P5, 64, 1, PW(ROW_GATE_B), nullptr, PIX_MSA, 64, CMc, 1.f, 2, 0);
  k_attn<<<dim3(16, 8, 128), 32, 0, stream>>>(P2, P3, P4, P5, nbbuf, msa_mask, P6, NNc, 8, 8, 0);
  gemm(stream, P6, 64, 1, PW(ROW_OUT_W), 64, 1, msa, 64, 1, PW(ROW_OUT_B), nullptr, PIX_MSA, 64, 64, 1.f, 0, 1);

  // ---------------- column global attention ----------------
  ln(stream, msa, PW(COL_LN_G), PW(COL_LN_B), P0, PIX_MSA, CMc);
  k_mean_m<<<NNc, 64, 0, stream>>>(P0, qavg, MM, NNc, CMc);
  gemm(stream, qavg, CMc, 1, PW(COL_Q), 64, 1, colq, 64, 1, nullptr, nullptr, NNc, 64, CMc, RS8, 0, 0);
  gemm(stream, P0, CMc, 1, PW(COL_K), 8, 1, P2, 8, 1, nullptr, nullptr, PIX_MSA, 8, CMc, 1.f, 0, 0);
  gemm(stream, P0, CMc, 1, PW(COL_V), 8, 1, P3, 8, 1, nullptr, nullptr, PIX_MSA, 8, CMc, 1.f, 0, 0);
  gemm(stream, P0, CMc, 1, PW(COL_GATE_W), 64, 1, P5, 64, 1, PW(COL_GATE_B), nullptr, PIX_MSA, 64, CMc, 1.f, 2, 0);
  k_col_attn<<<NNc, 128, 0, stream>>>(colq, P2, P3, P5, P6, MM, NNc);
  gemm(stream, P6, 64, 1, PW(COL_OUT_W), 64, 1, msa, 64, 1, PW(COL_OUT_B), nullptr, PIX_MSA, 64, 64, 1.f, 0, 1);

  // ---------------- msa transition ----------------
  ln(stream, msa, PW(MSATR_LN_G), PW(MSATR_LN_B), P0, PIX_MSA, CMc);
  k_transition<<<PIX_MSA / 16, 256, 0, stream>>>(P0, w1t_msa, PW(MSATR_B1),
                                                 w2t_msa, PW(MSATR_B2), msa,
                                                 (int)PIX_MSA, CMc, 256);

  // ---------------- outer product mean -> pair ----------------
  ln(stream, msa, PW(OPM_LN_G), PW(OPM_LN_B), P0, PIX_MSA, CMc);
  gemm(stream, P0, CMc, 1, PW(OPM_LW), 32, 1, P2, 32, 1, PW(OPM_LB), msa_mask, PIX_MSA, 32, CMc, 1.f, 0, 0);
  gemm(stream, P0, CMc, 1, PW(OPM_RW), 32, 1, P3, 32, 1, PW(OPM_RB), msa_mask, PIX_MSA, 32, CMc, 1.f, 0, 0);
  k_opm_norm<<<(PIX_PAIR + 255) / 256, 256, 0, stream>>>(msa_mask, nrm, MM, NNc);
  k_opm<<<PIX_PAIR / 16, 512, 0, stream>>>(P2, P3, owt, PW(OPM_OB), nrm, pair, NNc, MM);

  // ---------------- triangle multiplication (outgoing, then incoming) ----------------
  for (int dir = 0; dir < 2; ++dir) {
    const int LNg = dir ? TI_LN_G : TO_LN_G, LNb = dir ? TI_LN_B : TO_LN_B;
    const int LGw = dir ? TI_LG_W : TO_LG_W, LGb = dir ? TI_LG_B : TO_LG_B;
    const int LPw = dir ? TI_LP_W : TO_LP_W, LPb = dir ? TI_LP_B : TO_LP_B;
    const int RGw = dir ? TI_RG_W : TO_RG_W, RGb = dir ? TI_RG_B : TO_RG_B;
    const int RPw = dir ? TI_RP_W : TO_RP_W, RPb = dir ? TI_RP_B : TO_RP_B;
    const int Gw  = dir ? TI_G_W  : TO_G_W,  Gb  = dir ? TI_G_B  : TO_G_B;
    const int CLg = dir ? TI_CLN_G : TO_CLN_G, CLb = dir ? TI_CLN_B : TO_CLN_B;
    const int OPw = dir ? TI_OP_W : TO_OP_W, OPb = dir ? TI_OP_B : TO_OP_B;

    ln(stream, pair, PW(LNg), PW(LNb), P0, PIX_PAIR, CZc);
    gemm(stream, P0, CZc, 1, PW(LGw), 128, 1, P2, 128, 1, PW(LGb), nullptr, PIX_PAIR, 128, CZc, 1.f, 2, 0);
    gemm(stream, P0, CZc, 1, PW(LPw), 128, 1, P2, 128, 1, PW(LPb), pair_mask, PIX_PAIR, 128, CZc, 1.f, 0, 2);
    gemm(stream, P0, CZc, 1, PW(RGw), 128, 1, P3, 128, 1, PW(RGb), nullptr, PIX_PAIR, 128, CZc, 1.f, 2, 0);
    gemm(stream, P0, CZc, 1, PW(RPw), 128, 1, P3, 128, 1, PW(RPb), pair_mask, PIX_PAIR, 128, CZc, 1.f, 0, 2);
    gemm(stream, P0, CZc, 1, PW(Gw), 128, 1, P5, 128, 1, PW(Gb), nullptr, PIX_PAIR, 128, CZc, 1.f, 2, 0);
    const long RN = (long)NNc * CZc;   // 32768
    if (dir == 0) {   // outgoing: out[i,j,c] = sum_k lp[i,k,c] rp[j,k,c]
      gemm(stream, P2, RN, CZc, P3, CZc, RN, P4, RN, CZc, nullptr, nullptr,
           NNc, NNc, NNc, 1.f, 0, 0, CZc, 1, 1, 1);
    } else {          // incoming: out[i,j,c] = sum_k lp[k,j,c] rp[k,i,c]
      gemm(stream, P3, CZc, RN, P2, RN, CZc, P4, RN, CZc, nullptr, nullptr,
           NNc, NNc, NNc, 1.f, 0, 0, CZc, 1, 1, 1);
    }
    ln(stream, P4, PW(CLg), PW(CLb), P0, PIX_PAIR, CZc);
    gemm(stream, P0, CZc, 1, PW(OPw), 128, 1, P5, 128, 1, PW(OPb), nullptr, PIX_PAIR, 128, CZc, 1.f, 0, 2);
    k_add_tr<<<(PIX_PAIR * CZc + 255) / 256, 256, 0, stream>>>(pair, P5, NNc, CZc, 0);
  }

  // ---------------- triangle attention (starting, then ending) ----------------
  for (int ending = 0; ending < 2; ++ending) {
    const int B2D = ending ? TE_B2D : TS_B2D;
    const int Qw = ending ? TE_Q : TS_Q, Kw = ending ? TE_K : TS_K, Vw = ending ? TE_V : TS_V;
    const int Gw = ending ? TE_GATE_W : TS_GATE_W, Gb = ending ? TE_GATE_B : TS_GATE_B;
    const int Ow = ending ? TE_OUT_W : TS_OUT_W, Ob = ending ? TE_OUT_B : TS_OUT_B;
    const int LNg = ending ? TE_LN_G : TS_LN_G, LNb = ending ? TE_LN_B : TS_LN_B;

    ln(stream, pair, PW(LNg), PW(LNb), P0, PIX_PAIR, CZc);
    gemm(stream, P0, CZc, 1, PW(B2D), 4, 1, nbbuf, 1, LN2, nullptr, nullptr, PIX_PAIR, 4, CZc, 1.f, 0, 0);
    gemm(stream, P0, CZc, 1, PW(Qw), 128, 1, P2, 128, 1, nullptr, nullptr, PIX_PAIR, 128, CZc, RS32, 0, 0);
    gemm(stream, P0, CZc, 1, PW(Kw), 128, 1, P3, 128, 1, nullptr, nullptr, PIX_PAIR, 128, CZc, 1.f, 0, 0);
    gemm(stream, P0, CZc, 1, PW(Vw), 128, 1, P4, 128, 1, nullptr, nullptr, PIX_PAIR, 128, CZc, 1.f, 0, 0);
    gemm(stream, P0, CZc, 1, PW(Gw), 128, 1, P5, 128, 1, PW(Gb), nullptr, PIX_PAIR, 128, CZc, 1.f, 2, 0);
    k_attn<<<dim3(16, 4, 256), 32, 0, stream>>>(P2, P3, P4, P5, nbbuf, pair_mask, P6, NNc, 4, 32, ending);
    gemm(stream, P6, 128, 1, PW(Ow), 128, 1, P1, 128, 1, PW(Ob), nullptr, PIX_PAIR, 128, 128, 1.f, 0, 0);
    k_add_tr<<<(PIX_PAIR * CZc + 255) / 256, 256, 0, stream>>>(pair, P1, NNc, CZc, ending);
  }

  // ---------------- pair transition ----------------
  ln(stream, pair, PW(PTR_LN_G), PW(PTR_LN_B), P0, PIX_PAIR, CZc);
  k_transition<<<PIX_PAIR / 16, 256, 0, stream>>>(P0, w1t_pr, PW(PTR_B1),
                                                  w2t_pr, PW(PTR_B2), pair,
                                                  (int)PIX_PAIR, CZc, 512);
#undef PW
}
